// SparseMoE_14370960572799
// MI455X (gfx1250) — matrix operand order repaired and checked by reading, not compiled, not run on hardware
//
#include <hip/hip_runtime.h>
#include <cstdint>

#define T_TOK 8192
#define DIM_D 1024
#define DIM_H 2048
#define DIM_O 1024
#define NEXP  8
#define TOPK  2
#define BM 128
#define BN 128
#define BK 32
#define MAX_ROW_TILES (T_TOK * TOPK / BM + NEXP)   // 136

typedef __bf16 v16bf __attribute__((ext_vector_type(16)));
typedef float  v8f   __attribute__((ext_vector_type(8)));

union Frag { uint4 u[2]; v16bf v; };

static __device__ __forceinline__ unsigned short f2bf(float f) {
  unsigned u = __float_as_uint(f);
  unsigned r = (u + 0x7FFFu + ((u >> 16) & 1u)) >> 16;
  return (unsigned short)r;
}
static __device__ __forceinline__ float gelu_exact(float x) {
  return 0.5f * x * (1.0f + erff(x * 0.70710678118654752f));
}

// async global -> LDS copy, 16 bytes per lane (tracked by ASYNCcnt)
#define ASYNC_LD_B128(ldsaddr, gaddr)                                        \
  asm volatile("global_load_async_to_lds_b128 %0, %1, off"                   \
               :: "v"(ldsaddr), "v"(gaddr) : "memory")
#define WAIT_ASYNC0() asm volatile("s_wait_asynccnt 0x0" ::: "memory")

// ---------------- fp32 -> bf16 convert (vectorized) ----------------
__global__ __launch_bounds__(256) void cvt_bf16(const float4* __restrict__ src,
                                                unsigned short* __restrict__ dst,
                                                size_t n4) {
  size_t i = (size_t)blockIdx.x * blockDim.x + threadIdx.x;
  size_t stride = (size_t)gridDim.x * blockDim.x;
  for (; i < n4; i += stride) {
    float4 f = src[i];
    unsigned lo = (unsigned)f2bf(f.x) | ((unsigned)f2bf(f.y) << 16);
    unsigned hi = (unsigned)f2bf(f.z) | ((unsigned)f2bf(f.w) << 16);
    uint2 v; v.x = lo; v.y = hi;
    *(uint2*)(dst + 4 * i) = v;
  }
}

__global__ __launch_bounds__(256) void zero_f32(float* __restrict__ p, size_t n) {
  size_t i = (size_t)blockIdx.x * blockDim.x + threadIdx.x;
  size_t stride = (size_t)gridDim.x * blockDim.x;
  for (; i < n; i += stride) p[i] = 0.0f;
}

__global__ void zero_meta(int* __restrict__ counts) {
  if (threadIdx.x < NEXP) counts[threadIdx.x] = 0;
}

// ---------------- gating: one wave per token ----------------
__global__ __launch_bounds__(256) void gate_kernel(
    const float* __restrict__ x, const float* __restrict__ Wg,
    const float* __restrict__ bg, int* __restrict__ topk_idx,
    float* __restrict__ topk_w, int* __restrict__ counts) {
  int wid = threadIdx.x >> 5, lane = threadIdx.x & 31;
  int t = blockIdx.x * 8 + wid;
  if (t >= T_TOK) return;
  float acc[NEXP];
#pragma unroll
  for (int e = 0; e < NEXP; ++e) acc[e] = 0.0f;
  const float* xr = x + (size_t)t * DIM_D;
  for (int d = lane; d < DIM_D; d += 32) {
    float xv = xr[d];
#pragma unroll
    for (int e = 0; e < NEXP; ++e) acc[e] += xv * Wg[e * DIM_D + d];
  }
#pragma unroll
  for (int e = 0; e < NEXP; ++e) {
#pragma unroll
    for (int s = 16; s > 0; s >>= 1) acc[e] += __shfl_xor(acc[e], s, 32);
  }
  if (lane == 0) {
    float sc[NEXP];
#pragma unroll
    for (int e = 0; e < NEXP; ++e) sc[e] = acc[e] + bg[e];
    int i0 = 0;
#pragma unroll
    for (int e = 1; e < NEXP; ++e) if (sc[e] > sc[i0]) i0 = e;   // lowest index on ties
    int i1 = (i0 == 0) ? 1 : 0;
#pragma unroll
    for (int e = 0; e < NEXP; ++e) if (e != i0 && sc[e] > sc[i1]) i1 = e;
    float e1 = __expf(sc[i1] - sc[i0]);      // sc[i0] >= sc[i1]
    float inv = 1.0f / (1.0f + e1);
    topk_idx[t * 2 + 0] = i0; topk_idx[t * 2 + 1] = i1;
    topk_w[t * 2 + 0] = inv;  topk_w[t * 2 + 1] = e1 * inv;
    atomicAdd(&counts[i0], 1); atomicAdd(&counts[i1], 1);
  }
}

// ---------------- tiny scan: offsets / tile offsets ----------------
__global__ void scan_kernel(const int* __restrict__ counts, int* __restrict__ offsets,
                            int* __restrict__ tileOff, int* __restrict__ cursor) {
  if (threadIdx.x == 0) {
    int o = 0, to = 0;
    for (int e = 0; e < NEXP; ++e) {
      offsets[e] = o; tileOff[e] = to; cursor[e] = 0;
      int c = counts[e];
      o += c; to += (c + BM - 1) / BM;
    }
    offsets[NEXP] = o; tileOff[NEXP] = to;
  }
}

// ---------------- placement: compact per-expert assignment lists ----------------
__global__ __launch_bounds__(256) void place_kernel(
    const int* __restrict__ topk_idx, const float* __restrict__ topk_w,
    const int* __restrict__ offsets, int* __restrict__ cursor,
    int* __restrict__ perm_tok, float* __restrict__ perm_w) {
  int t = blockIdx.x * 256 + threadIdx.x;
  if (t >= T_TOK) return;
#pragma unroll
  for (int k = 0; k < TOPK; ++k) {
    int e = topk_idx[t * 2 + k];
    int pos = offsets[e] + atomicAdd(&cursor[e], 1);
    perm_tok[pos] = t;
    perm_w[pos] = topk_w[t * 2 + k];
  }
}

// ---------------- grouped GEMM with WMMA bf16, async double-buffered LDS ----------------
// MODE 0 (fc1): A = xb gathered via perm, B = W1b[e] [H][D]; epilogue +b1, GELU -> hbuf bf16
// MODE 1 (fc2): A = hbuf rows,             B = W2b[e] [O][H]; epilogue w*(acc+b2) -> atomicAdd out
template <int MODE>
__global__ __launch_bounds__(256) void moe_gemm(
    const unsigned short* __restrict__ Abase, const unsigned short* __restrict__ Bbase,
    const float* __restrict__ bias, const int* __restrict__ offsets,
    const int* __restrict__ tileOff, const int* __restrict__ perm_tok,
    const float* __restrict__ perm_w, unsigned short* __restrict__ hbuf,
    float* __restrict__ out, int Kdim, int N) {
  int bx = blockIdx.x;
  if (bx >= tileOff[NEXP]) return;
  int e = 0;
  while (e < NEXP - 1 && bx >= tileOff[e + 1]) ++e;
  int row0 = offsets[e] + (bx - tileOff[e]) * BM;
  int rows = offsets[e + 1] - row0; if (rows > BM) rows = BM;
  int n0 = blockIdx.y * BN;

  __shared__ __align__(16) unsigned short As[2][BM * BK];
  __shared__ __align__(16) unsigned short Bs[2][BN * BK];

  int tid = threadIdx.x;
  int lane = tid & 31, wid = tid >> 5;
  int wm = wid >> 1, wn = wid & 1;       // 4 x 2 wave grid
  int hh = lane >> 4, l16 = lane & 15;   // half / lane-in-16

  const unsigned short* Bexp = Bbase + (size_t)e * (size_t)N * (size_t)Kdim;

  // Per-thread copy chunks: tile = 128 rows x 32 cols bf16 = 512 uint4, 2 per thread.
  unsigned ldsA[2][2], ldsB[2][2];
  unsigned long long gA[2], gB[2];
#pragma unroll
  for (int p = 0; p < 2; ++p) {
    int chunk = p * 256 + tid;          // [0, 512)
    int r = chunk >> 2;                 // [0, 128)
    int c = (chunk & 3) * 8;            // 0,8,16,24
    int rr = (r < rows) ? r : 0;
    int arow = (MODE == 0) ? perm_tok[row0 + rr] : (row0 + rr);
    gA[p] = (unsigned long long)(Abase + (size_t)arow * Kdim + c);
    gB[p] = (unsigned long long)(Bexp + (size_t)(n0 + r) * Kdim + c);
#pragma unroll
    for (int b = 0; b < 2; ++b) {
      ldsA[b][p] = (unsigned)(uintptr_t)&As[b][r * BK + c];
      ldsB[b][p] = (unsigned)(uintptr_t)&Bs[b][r * BK + c];
    }
  }

  v8f acc[2][4];
#pragma unroll
  for (int mi = 0; mi < 2; ++mi)
#pragma unroll
    for (int ni = 0; ni < 4; ++ni)
#pragma unroll
      for (int r = 0; r < 8; ++r) acc[mi][ni][r] = 0.0f;

  // prologue: async-fill buffer 0 with the k0 = 0 tiles
#pragma unroll
  for (int p = 0; p < 2; ++p) {
    ASYNC_LD_B128(ldsA[0][p], gA[p]);
    ASYNC_LD_B128(ldsB[0][p], gB[p]);
  }

  int nIter = Kdim / BK;
  for (int it = 0; it < nIter; ++it) {
    int buf = it & 1;
    WAIT_ASYNC0();        // my async copies into `buf` are in LDS
    __syncthreads();      // everyone's copies landed; everyone done reading buf^1

    if (it + 1 < nIter) { // stream next K-tile into the other buffer during WMMAs
      unsigned long long koff = (unsigned long long)(it + 1) * (BK * 2);
#pragma unroll
      for (int p = 0; p < 2; ++p) {
        ASYNC_LD_B128(ldsA[buf ^ 1][p], gA[p] + koff);
        ASYNC_LD_B128(ldsB[buf ^ 1][p], gB[p] + koff);
      }
    }

    Frag a[2], b[4];
#pragma unroll
    for (int mi = 0; mi < 2; ++mi) {
      int m = wm * 32 + mi * 16 + l16;
      a[mi].u[0] = *(const uint4*)(&As[buf][m * BK + 8 * hh]);        // K = 8h + 0..7
      a[mi].u[1] = *(const uint4*)(&As[buf][m * BK + 16 + 8 * hh]);   // K = 16 + 8h + 0..7
    }
#pragma unroll
    for (int ni = 0; ni < 4; ++ni) {
      int n = wn * 64 + ni * 16 + l16;
      b[ni].u[0] = *(const uint4*)(&Bs[buf][n * BK + 8 * hh]);        // K = 8h + 0..7
      b[ni].u[1] = *(const uint4*)(&Bs[buf][n * BK + 16 + 8 * hh]);   // K = 16 + 8h + 0..7
    }
#pragma unroll
    for (int mi = 0; mi < 2; ++mi)
#pragma unroll
      for (int ni = 0; ni < 4; ++ni)
        acc[mi][ni] = __builtin_amdgcn_wmma_f32_16x16x32_bf16(
            false, a[mi].v, false, b[ni].v, (short)0, acc[mi][ni], false, false);
  }

  // epilogue  (C layout: VGPR r, M = r + 8*half, N = lane&15)
#pragma unroll
  for (int mi = 0; mi < 2; ++mi) {
#pragma unroll
    for (int ni = 0; ni < 4; ++ni) {
      int ncol = n0 + wn * 64 + ni * 16 + l16;
      float bval = bias[(size_t)e * N + ncol];
#pragma unroll
      for (int r = 0; r < 8; ++r) {
        int mrow = wm * 32 + mi * 16 + 8 * hh + r;
        if (mrow < rows) {
          int g = row0 + mrow;
          float v = acc[mi][ni][r] + bval;
          if (MODE == 0) {
            hbuf[(size_t)g * N + ncol] = f2bf(gelu_exact(v));
          } else {
            int t = perm_tok[g];
            float w = perm_w[g];
            atomicAdd(&out[(size_t)t * N + ncol], w * v);
          }
        }
      }
    }
  }
}

// ---------------- host launcher ----------------
extern "C" void kernel_launch(void* const* d_in, const int* in_sizes, int n_in,
                              void* d_out, int out_size, void* d_ws, size_t ws_size,
                              hipStream_t stream) {
  (void)in_sizes; (void)n_in; (void)out_size; (void)ws_size;
  const float* x  = (const float*)d_in[0];
  const float* Wg = (const float*)d_in[1];
  const float* bg = (const float*)d_in[2];
  const float* W1 = (const float*)d_in[3];
  const float* b1 = (const float*)d_in[4];
  const float* W2 = (const float*)d_in[5];
  const float* b2 = (const float*)d_in[6];
  float* out = (float*)d_out;

  uintptr_t p = (uintptr_t)d_ws;
  auto take = [&](size_t bytes) -> void* {
    uintptr_t cur = (p + 255) & ~(uintptr_t)255;
    p = cur + bytes;
    return (void*)cur;
  };
  unsigned short* xb   = (unsigned short*)take((size_t)T_TOK * DIM_D * 2);
  unsigned short* W1b  = (unsigned short*)take((size_t)NEXP * DIM_H * DIM_D * 2);
  unsigned short* W2b  = (unsigned short*)take((size_t)NEXP * DIM_O * DIM_H * 2);
  unsigned short* hbuf = (unsigned short*)take((size_t)T_TOK * TOPK * DIM_H * 2);
  int*   topk_idx = (int*)  take((size_t)T_TOK * TOPK * 4);
  float* topk_w   = (float*)take((size_t)T_TOK * TOPK * 4);
  int*   perm_tok = (int*)  take((size_t)T_TOK * TOPK * 4);
  float* perm_w   = (float*)take((size_t)T_TOK * TOPK * 4);
  int* counts  = (int*)take(64);
  int* cursor  = (int*)take(64);
  int* offsets = (int*)take(64);
  int* tileOff = (int*)take(64);

  zero_meta<<<1, 32, 0, stream>>>(counts);

  cvt_bf16<<<2048, 256, 0, stream>>>((const float4*)x,  xb,  (size_t)T_TOK * DIM_D / 4);
  cvt_bf16<<<2048, 256, 0, stream>>>((const float4*)W1, W1b, (size_t)NEXP * DIM_H * DIM_D / 4);
  cvt_bf16<<<2048, 256, 0, stream>>>((const float4*)W2, W2b, (size_t)NEXP * DIM_O * DIM_H / 4);

  gate_kernel<<<T_TOK / 8, 256, 0, stream>>>(x, Wg, bg, topk_idx, topk_w, counts);
  scan_kernel<<<1, 32, 0, stream>>>(counts, offsets, tileOff, cursor);
  place_kernel<<<T_TOK / 256, 256, 0, stream>>>(topk_idx, topk_w, offsets, cursor,
                                                perm_tok, perm_w);

  // fc1: [rows x D] x [D x H] -> GELU -> hbuf
  moe_gemm<0><<<dim3(MAX_ROW_TILES, DIM_H / BN), 256, 0, stream>>>(
      xb, W1b, b1, offsets, tileOff, perm_tok, perm_w, hbuf, nullptr, DIM_D, DIM_H);

  zero_f32<<<2048, 256, 0, stream>>>(out, (size_t)T_TOK * DIM_O);

  // fc2: [rows x H] x [H x O] -> weighted scatter-add into out
  moe_gemm<1><<<dim3(MAX_ROW_TILES, DIM_O / BN), 256, 0, stream>>>(
      hbuf, W2b, b2, offsets, tileOff, perm_tok, perm_w, nullptr, out, DIM_H, DIM_O);
}
